// Z2allAttention_11192684773857
// MI455X (gfx1250) — matrix operand order repaired and checked by reading, not compiled, not run on hardware
//
#include <hip/hip_runtime.h>

typedef __attribute__((ext_vector_type(16))) __bf16 v16bf;
typedef __attribute__((ext_vector_type(8)))  float  v8f;

#define B_  2
#define S_  2048
#define H_  1024
#define NH_ 16
#define HD_ 64

union FragBF { v16bf v; uint4 q[2]; };

// ---------------------------------------------------------------------------
// Fragment loads (global or LDS source; addrspace inferred after inlining)
// A-fragment (16x32, M x K): lane l: m=l&15, chunks at k0+(l>>4)*8 and +16
// ---------------------------------------------------------------------------
__device__ __forceinline__ v16bf load_a_frag(const __bf16* base, int ld, int m0, int k0, int lane) {
  int m  = m0 + (lane & 15);
  int kh = (lane >> 4) * 8;
  const __bf16* p = base + (size_t)m * ld + (k0 + kh);
  FragBF f;
  f.q[0] = *(const uint4*)(p);
  f.q[1] = *(const uint4*)(p + 16);
  return f.v;
}

// B-fragment (32x16, K x N) with B[k,n] = src[n,k] (row-major src, chunks at k0+(l>>4)*8 and +16)
__device__ __forceinline__ v16bf load_b_frag(const __bf16* base, int ld, int n0, int k0, int lane) {
  int n  = n0 + (lane & 15);
  int kh = (lane >> 4) * 8;
  const __bf16* p = base + (size_t)n * ld + (k0 + kh);
  FragBF f;
  f.q[0] = *(const uint4*)(p);
  f.q[1] = *(const uint4*)(p + 16);
  return f.v;
}

__device__ __forceinline__ float lane16_max(float v) {
  v = fmaxf(v, __shfl_xor(v, 1, 32));
  v = fmaxf(v, __shfl_xor(v, 2, 32));
  v = fmaxf(v, __shfl_xor(v, 4, 32));
  v = fmaxf(v, __shfl_xor(v, 8, 32));
  return v;
}
__device__ __forceinline__ float lane16_sum(float v) {
  v += __shfl_xor(v, 1, 32);
  v += __shfl_xor(v, 2, 32);
  v += __shfl_xor(v, 4, 32);
  v += __shfl_xor(v, 8, 32);
  return v;
}

// ---------------------------------------------------------------------------
// CDNA5 async global->LDS copy (ASYNCcnt-tracked), per CDNA5 ISA
// ---------------------------------------------------------------------------
__device__ __forceinline__ void async_copy_b128(uint32_t lds_off, const __bf16* g) {
  asm volatile("global_load_async_to_lds_b128 %0, %1, off"
               :: "v"(lds_off), "v"(g)
               : "memory");
}
__device__ __forceinline__ void wait_async_le(int n) {
  if (n == 0) asm volatile("s_wait_asynccnt 0x0" ::: "memory");
  else        asm volatile("s_wait_asynccnt 0x4" ::: "memory");
}

// ---------------------------------------------------------------------------
// f32 -> bf16 conversion
// ---------------------------------------------------------------------------
__global__ void f32_to_bf16(const float* __restrict__ src, __bf16* __restrict__ dst, int n) {
  int i = blockIdx.x * blockDim.x + threadIdx.x;
  if (i < n) dst[i] = (__bf16)src[i];
}

// ---------------------------------------------------------------------------
// GEMM: C[m,n] = sum_k A[m,k] * W[n,k]  (A: [4096,1024] bf16, W: [1024,1024] bf16)
// 256 threads = 8 waves; workgroup tile 128x128; wave tile 32x64; K step 32.
// Operands staged global->LDS with async-to-LDS (double-buffered, 32 KB).
// OUT_MODE 0: f32 row-major [4096,1024]
// OUT_MODE 2: bf16 [B,NH,S,HD]
// OUT_MODE 3: bf16 [B,NH,HD,S]   (transposed, for attention V)
// ---------------------------------------------------------------------------
template<int OUT_MODE>
__global__ __launch_bounds__(256)
void gemm_wmma_bf16(const __bf16* __restrict__ A, const __bf16* __restrict__ W,
                    void* __restrict__ Cout) {
  __shared__ __align__(16) __bf16 Atile[2][128 * 32];
  __shared__ __align__(16) __bf16 Btile[2][128 * 32];

  const int lane = threadIdx.x & 31;
  const int wid  = threadIdx.x >> 5;
  const int wm = wid & 3, wn = wid >> 2;
  const int m0 = blockIdx.y * 128;
  const int n0 = blockIdx.x * 128;

  // Each thread copies two 16B chunks of the A tile and two of the B tile per step.
  // Tile = 128 rows x 32 k (bf16) = 512 chunks of 16B; chunk c: row=c>>2, sub=c&3.
  const int t   = threadIdx.x;
  const int c0  = t,        c1  = t + 256;
  const int rA0 = c0 >> 2,  sA0 = c0 & 3;
  const int rA1 = c1 >> 2,  sA1 = c1 & 3;
  const __bf16* gA0 = A + (size_t)(m0 + rA0) * H_ + sA0 * 8;
  const __bf16* gA1 = A + (size_t)(m0 + rA1) * H_ + sA1 * 8;
  const __bf16* gB0 = W + (size_t)(n0 + rA0) * H_ + sA0 * 8;
  const __bf16* gB1 = W + (size_t)(n0 + rA1) * H_ + sA1 * 8;
  uint32_t lA0[2], lA1[2], lB0[2], lB1[2];
  #pragma unroll
  for (int buf = 0; buf < 2; ++buf) {
    lA0[buf] = (uint32_t)(uintptr_t)&Atile[buf][c0 * 8];
    lA1[buf] = (uint32_t)(uintptr_t)&Atile[buf][c1 * 8];
    lB0[buf] = (uint32_t)(uintptr_t)&Btile[buf][c0 * 8];
    lB1[buf] = (uint32_t)(uintptr_t)&Btile[buf][c1 * 8];
  }

  v8f acc[2][4] = {};

  // Prologue: stage k-step 0 into buffer 0 (4 async instructions per wave).
  async_copy_b128(lA0[0], gA0);
  async_copy_b128(lA1[0], gA1);
  async_copy_b128(lB0[0], gB0);
  async_copy_b128(lB1[0], gB1);

  const int NSTEP = H_ / 32;
  for (int i = 0; i < NSTEP; ++i) {
    const int cur = i & 1;
    if (i + 1 < NSTEP) {
      const int k1 = (i + 1) * 32;
      async_copy_b128(lA0[1 - cur], gA0 + k1);
      async_copy_b128(lA1[1 - cur], gA1 + k1);
      async_copy_b128(lB0[1 - cur], gB0 + k1);
      async_copy_b128(lB1[1 - cur], gB1 + k1);
      wait_async_le(4);          // in-order completion => current buffer landed
    } else {
      wait_async_le(0);
    }
    __syncthreads();             // whole tile visible to all waves

    const __bf16* At = Atile[cur];
    const __bf16* Bt = Btile[cur];
    v16bf a0 = load_a_frag(At, 32, wm * 32,      0, lane);
    v16bf a1 = load_a_frag(At, 32, wm * 32 + 16, 0, lane);
    v16bf b0 = load_b_frag(Bt, 32, wn * 64,      0, lane);
    v16bf b1 = load_b_frag(Bt, 32, wn * 64 + 16, 0, lane);
    v16bf b2 = load_b_frag(Bt, 32, wn * 64 + 32, 0, lane);
    v16bf b3 = load_b_frag(Bt, 32, wn * 64 + 48, 0, lane);
    __syncthreads();             // all reads done before buffer is refilled later

    acc[0][0] = __builtin_amdgcn_wmma_f32_16x16x32_bf16(false, a0, false, b0, (short)0, acc[0][0], false, false);
    acc[0][1] = __builtin_amdgcn_wmma_f32_16x16x32_bf16(false, a0, false, b1, (short)0, acc[0][1], false, false);
    acc[0][2] = __builtin_amdgcn_wmma_f32_16x16x32_bf16(false, a0, false, b2, (short)0, acc[0][2], false, false);
    acc[0][3] = __builtin_amdgcn_wmma_f32_16x16x32_bf16(false, a0, false, b3, (short)0, acc[0][3], false, false);
    acc[1][0] = __builtin_amdgcn_wmma_f32_16x16x32_bf16(false, a1, false, b0, (short)0, acc[1][0], false, false);
    acc[1][1] = __builtin_amdgcn_wmma_f32_16x16x32_bf16(false, a1, false, b1, (short)0, acc[1][1], false, false);
    acc[1][2] = __builtin_amdgcn_wmma_f32_16x16x32_bf16(false, a1, false, b2, (short)0, acc[1][2], false, false);
    acc[1][3] = __builtin_amdgcn_wmma_f32_16x16x32_bf16(false, a1, false, b3, (short)0, acc[1][3], false, false);
  }

  // Epilogue. C layout: lane l holds column n=l&15, rows (l>>4)*8 + r.
  const int wm0 = m0 + wm * 32;
  const int wn0 = n0 + wn * 64;
  #pragma unroll
  for (int i = 0; i < 2; ++i) {
    #pragma unroll
    for (int j = 0; j < 4; ++j) {
      #pragma unroll
      for (int r = 0; r < 8; ++r) {
        int gm = wm0 + i * 16 + (lane >> 4) * 8 + r;
        int gn = wn0 + j * 16 + (lane & 15);
        float val = acc[i][j][r];
        if (OUT_MODE == 0) {
          ((float*)Cout)[(size_t)gm * H_ + gn] = val;
        } else if (OUT_MODE == 2) {
          int b = gm >> 11, s = gm & (S_ - 1), h = gn >> 6, d = gn & (HD_ - 1);
          ((__bf16*)Cout)[(((size_t)(b * NH_ + h)) * S_ + s) * HD_ + d] = (__bf16)val;
        } else {
          int b = gm >> 11, s = gm & (S_ - 1), h = gn >> 6, d = gn & (HD_ - 1);
          ((__bf16*)Cout)[(((size_t)(b * NH_ + h)) * HD_ + d) * S_ + s] = (__bf16)val;
        }
      }
    }
  }
}

// ---------------------------------------------------------------------------
// RoPE (interleaved pairs), faithful to the reference bug:
// k' uses rotate_half(q), not rotate_half(k). In-place on [B,NH,S,HD] bf16.
// ---------------------------------------------------------------------------
__global__ void rope_kernel(__bf16* __restrict__ Qb, __bf16* __restrict__ Kb,
                            const float* __restrict__ cosT, const float* __restrict__ sinT,
                            const int* __restrict__ pos) {
  int idx = blockIdx.x * blockDim.x + threadIdx.x;           // 0 .. 2^21-1
  int j  = idx & 31;
  int s  = (idx >> 5) & (S_ - 1);
  int bh = idx >> 16;                                        // 0 .. 31
  if (bh >= B_ * NH_) return;
  size_t base = (((size_t)bh * S_) + s) * HD_ + j * 2;
  int p = pos[s];
  float c  = cosT[p * HD_ + 2 * j];
  float sn = sinT[p * HD_ + 2 * j];
  float q0 = (float)Qb[base], q1 = (float)Qb[base + 1];
  float k0 = (float)Kb[base], k1 = (float)Kb[base + 1];
  Qb[base]     = (__bf16)(q0 * c - q1 * sn);
  Qb[base + 1] = (__bf16)(q1 * c + q0 * sn);
  Kb[base]     = (__bf16)(k0 * c - q1 * sn);   // bug preserved: rotate_half(q)
  Kb[base + 1] = (__bf16)(k1 * c + q0 * sn);
}

// ---------------------------------------------------------------------------
// Flash attention: 4 waves x 16 queries = 64-query tile, 32-key blocks,
// online softmax, WMMA for QK^T and PV. V is pre-transposed [B,NH,HD,S].
// Output: bf16 row-major [B*S, H].
// ---------------------------------------------------------------------------
__global__ __launch_bounds__(128)
void attn_wmma(const __bf16* __restrict__ Q, const __bf16* __restrict__ K,
               const __bf16* __restrict__ Vt, __bf16* __restrict__ Out) {
  __shared__ __align__(16) __bf16 lds_p[4][16 * 32];

  const int lane = threadIdx.x & 31;
  const int wid  = threadIdx.x >> 5;
  const int b = blockIdx.z, h = blockIdx.y;
  const int q0 = blockIdx.x * 64 + wid * 16;
  const size_t bh = (size_t)(b * NH_ + h);
  const __bf16* Qbh = Q  + bh * S_ * HD_;
  const __bf16* Kbh = K  + bh * S_ * HD_;
  const __bf16* Vbh = Vt + bh * HD_ * S_;

  v16bf qa0 = load_a_frag(Qbh, HD_, q0, 0,  lane);
  v16bf qa1 = load_a_frag(Qbh, HD_, q0, 32, lane);

  v8f acc[4] = {};
  float mrow[8], lrow[8];
  #pragma unroll
  for (int r = 0; r < 8; ++r) { mrow[r] = -1e30f; lrow[r] = 0.0f; }

  const int rowbase = q0 + (lane >> 4) * 8;
  const int col = lane & 15;
  const int kend = blockIdx.x * 64 + 64;    // causal bound for the whole tile

  for (int kb = 0; kb < kend; kb += 32) {
    // ---- scores: S = Q * K^T (16 x 32 per wave) ----
    v8f sc[2] = {};
    {
      v16bf bk00 = load_b_frag(Kbh, HD_, kb,      0,  lane);
      v16bf bk01 = load_b_frag(Kbh, HD_, kb,      32, lane);
      v16bf bk10 = load_b_frag(Kbh, HD_, kb + 16, 0,  lane);
      v16bf bk11 = load_b_frag(Kbh, HD_, kb + 16, 32, lane);
      sc[0] = __builtin_amdgcn_wmma_f32_16x16x32_bf16(false, qa0, false, bk00, (short)0, sc[0], false, false);
      sc[0] = __builtin_amdgcn_wmma_f32_16x16x32_bf16(false, qa1, false, bk01, (short)0, sc[0], false, false);
      sc[1] = __builtin_amdgcn_wmma_f32_16x16x32_bf16(false, qa0, false, bk10, (short)0, sc[1], false, false);
      sc[1] = __builtin_amdgcn_wmma_f32_16x16x32_bf16(false, qa1, false, bk11, (short)0, sc[1], false, false);
    }

    // ---- scale + causal mask + online softmax (per row r) ----
    #pragma unroll
    for (int r = 0; r < 8; ++r) {
      int row = rowbase + r;
      float s0 = sc[0][r] * 0.125f;     // 1/sqrt(64)
      float s1 = sc[1][r] * 0.125f;
      if (kb + col > row)      s0 = -1e30f;
      if (kb + 16 + col > row) s1 = -1e30f;
      float tmax  = lane16_max(fmaxf(s0, s1));
      float mnew  = fmaxf(mrow[r], tmax);
      float alpha = __expf(mrow[r] - mnew);
      float p0 = __expf(s0 - mnew);
      float p1 = __expf(s1 - mnew);
      lrow[r] = lrow[r] * alpha + lane16_sum(p0 + p1);
      mrow[r] = mnew;
      #pragma unroll
      for (int dt = 0; dt < 4; ++dt) acc[dt][r] *= alpha;
      sc[0][r] = p0; sc[1][r] = p1;
    }

    // ---- P: C-layout -> A-layout via per-wave LDS tile (16x32 bf16) ----
    __bf16* pt = lds_p[wid];
    const int rb = (lane >> 4) * 8;
    #pragma unroll
    for (int r = 0; r < 8; ++r) {
      pt[(rb + r) * 32 + col]      = (__bf16)sc[0][r];
      pt[(rb + r) * 32 + 16 + col] = (__bf16)sc[1][r];
    }
    __syncthreads();
    v16bf pa;
    {
      const __bf16* p = pt + (lane & 15) * 32 + (lane >> 4) * 8;
      FragBF f;
      f.q[0] = *(const uint4*)(p);
      f.q[1] = *(const uint4*)(p + 16);
      pa = f.v;
    }
    __syncthreads();

    // ---- O += P * V  (V transposed: B[k,n] = Vt[d, s]) ----
    #pragma unroll
    for (int dt = 0; dt < 4; ++dt) {
      v16bf bv = load_b_frag(Vbh, S_, dt * 16, kb, lane);
      acc[dt] = __builtin_amdgcn_wmma_f32_16x16x32_bf16(false, pa, false, bv, (short)0, acc[dt], false, false);
    }
  }

  // ---- normalize and write bf16 [B*S, H] ----
  #pragma unroll
  for (int dt = 0; dt < 4; ++dt) {
    #pragma unroll
    for (int r = 0; r < 8; ++r) {
      float o = acc[dt][r] / lrow[r];
      int row = rowbase + r;
      int c   = h * HD_ + dt * 16 + col;
      Out[((size_t)b * S_ + row) * H_ + c] = (__bf16)o;
    }
  }
}

// ---------------------------------------------------------------------------
// Launch
// ---------------------------------------------------------------------------
extern "C" void kernel_launch(void* const* d_in, const int* in_sizes, int n_in,
                              void* d_out, int out_size, void* d_ws, size_t ws_size,
                              hipStream_t stream) {
  const float* hidden = (const float*)d_in[0];
  const float* fcos   = (const float*)d_in[1];
  const float* fsin   = (const float*)d_in[2];
  const int*   pos    = (const int*)d_in[3];
  const float* Wq     = (const float*)d_in[4];
  const float* Wk     = (const float*)d_in[5];
  const float* Wv     = (const float*)d_in[6];
  const float* Wo     = (const float*)d_in[7];

  char* ws = (char*)d_ws;
  const size_t MB = 1024 * 1024;
  __bf16* Xb  = (__bf16*)(ws);             // 8 MB  : X bf16 [4096,1024]
  __bf16* Wqb = (__bf16*)(ws + 8  * MB);   // 2 MB
  __bf16* Wkb = (__bf16*)(ws + 10 * MB);   // 2 MB
  __bf16* Wvb = (__bf16*)(ws + 12 * MB);   // 2 MB
  __bf16* Wob = (__bf16*)(ws + 14 * MB);   // 2 MB
  __bf16* Qb  = (__bf16*)(ws + 16 * MB);   // 8 MB  : [B,NH,S,HD]
  __bf16* Kb  = (__bf16*)(ws + 24 * MB);   // 8 MB  : [B,NH,S,HD]
  __bf16* Vb  = (__bf16*)(ws + 32 * MB);   // 8 MB  : [B,NH,HD,S]
  __bf16* AOb = (__bf16*)(ws + 40 * MB);   // 8 MB  : [B*S, H]

  const int nX = B_ * S_ * H_;
  const int nW = H_ * H_;
  f32_to_bf16<<<(nX + 255) / 256, 256, 0, stream>>>(hidden, Xb, nX);
  f32_to_bf16<<<(nW + 255) / 256, 256, 0, stream>>>(Wq, Wqb, nW);
  f32_to_bf16<<<(nW + 255) / 256, 256, 0, stream>>>(Wk, Wkb, nW);
  f32_to_bf16<<<(nW + 255) / 256, 256, 0, stream>>>(Wv, Wvb, nW);
  f32_to_bf16<<<(nW + 255) / 256, 256, 0, stream>>>(Wo, Wob, nW);

  dim3 ggrid(H_ / 128, (B_ * S_) / 128);   // (8, 32)
  gemm_wmma_bf16<2><<<ggrid, 256, 0, stream>>>(Xb, Wqb, Qb);
  gemm_wmma_bf16<2><<<ggrid, 256, 0, stream>>>(Xb, Wkb, Kb);
  gemm_wmma_bf16<3><<<ggrid, 256, 0, stream>>>(Xb, Wvb, Vb);

  const int nPairs = B_ * NH_ * S_ * (HD_ / 2);
  rope_kernel<<<(nPairs + 255) / 256, 256, 0, stream>>>(Qb, Kb, fcos, fsin, pos);

  dim3 agrid(S_ / 64, NH_, B_);
  attn_wmma<<<agrid, 128, 0, stream>>>(Qb, Kb, Vb, AOb);

  gemm_wmma_bf16<0><<<ggrid, 256, 0, stream>>>(AOb, Wob, (float*)d_out);
}
